// GraphCastModel_69621419868958
// MI455X (gfx1250) — compile-verified, matrix operand order checked by reading, not run on hardware
//
#include <hip/hip_runtime.h>
#include <hip/hip_bf16.h>

// ---------------- problem constants ----------------
#define BB      2
#define VV      78
#define NGRID   65160      // 181*360
#define NMESH   10242
#define NEDGE   163840
#define NODE_D  256
#define EDGE_D  64
#define HIDD    512
#define NLAYERS 16

typedef __attribute__((ext_vector_type(16))) _Float16 v16h;
typedef __attribute__((ext_vector_type(8)))  float    v8f;

static __device__ __forceinline__ int imin(int a, int b) { return a < b ? a : b; }

// ---------------- CDNA5 async global->LDS copy (probe-verified builtin) ----------------
#if defined(__HIP_DEVICE_COMPILE__) && __has_builtin(__builtin_amdgcn_global_load_async_to_lds_b128)
#define HAVE_ASYNC_LDS 1
#endif

typedef int av4i __attribute__((vector_size(16)));

static __device__ __forceinline__ void cp16_g2lds(_Float16* lds, const _Float16* g) {
#if defined(HAVE_ASYNC_LDS)
  __builtin_amdgcn_global_load_async_to_lds_b128(
      (__attribute__((address_space(1))) av4i*)g,
      (__attribute__((address_space(3))) av4i*)lds, 0, 0);
#else
  *(uint4*)lds = *(const uint4*)g;
#endif
}

static __device__ __forceinline__ void async_wait() {
#if defined(HAVE_ASYNC_LDS)
#if __has_builtin(__builtin_amdgcn_s_wait_asynccnt)
  __builtin_amdgcn_s_wait_asynccnt(0);
#else
  asm volatile("s_wait_asynccnt 0x0" ::: "memory");
#endif
#endif
}

// ---------------- fragment loaders ----------------
// A fragment (16x32 f16, M x K) from LDS, row-major, stride in halfs.
// Per CDNA5 ISA: lanes 0-15 rows M=0..15 Kbase 0, lanes 16-31 rows M=0..15 Kbase 8;
// halfs 0..7 -> K kb..kb+7, halfs 8..15 -> K kb+16..kb+23.
static __device__ __forceinline__ v16h load_a_frag(const _Float16* x, int strideHalfs, int kTile) {
  const int lane = threadIdx.x & 31;
  const int m    = lane & 15;
  const int kb   = kTile * 32 + ((lane & 16) ? 8 : 0);
  const _Float16* p = x + (size_t)m * strideHalfs + kb;
  union { v16h v; uint4 q[2]; } u;
  u.q[0] = *(const uint4*)(p);
  u.q[1] = *(const uint4*)(p + 16);
  return u.v;
}

// B fragment (32x16 f16, K x N), pre-packed contiguously: frag = 32 lanes x 16 halfs.
static __device__ __forceinline__ v16h load_b_frag(const _Float16* p) {
  union { v16h v; uint4 q[2]; } u;
  u.q[0] = *(const uint4*)(p);
  u.q[1] = *(const uint4*)(p + 8);
  return u.v;
}

#define WMMA(a, b, c) \
  __builtin_amdgcn_wmma_f32_16x16x32_f16(false, (a), false, (b), (short)0, (c), false, false)

// ---------------- fused two-layer MLP core, 64-row block ----------------
// Y[64 x OUTP] = relu(X @ W1 + b1) @ W2 + b2 (+ optional residual from LDS X cols 0..OUT-1)
// Block = 256 threads (8 wave32). ldsX: 64 x KPAD f16. ldsH: 64 x 512 f16.
template <int KPAD, int OUTP>
static __device__ __forceinline__ void mlp_core(
    const _Float16* ldsX, _Float16* ldsH,
    const _Float16* __restrict__ w1, const float* __restrict__ b1,
    const _Float16* __restrict__ w2, const float* __restrict__ b2,
    void* out, int outStride, bool outIsF16,
    int validRows, int validCols, bool residual)
{
  const int wave    = threadIdx.x >> 5;
  const int lane    = threadIdx.x & 31;
  const int halfSel = (lane & 16) ? 8 : 0;
  const int nSub    = lane & 15;
  constexpr int KT1 = KPAD / 32;

  // ---- GEMM1: hidden = relu(X @ W1 + b1) -> LDS f16 (64 x 512) ----
  {
    const int rt = wave >> 1;                 // row tile 0..3
    const int ch = wave & 1;                  // column half (16 col-tiles each)
    const _Float16* Xrt = ldsX + (size_t)rt * 16 * KPAD;
    for (int pass = 0; pass < 2; ++pass) {
      v8f acc[8] = {};
      for (int k = 0; k < KT1; ++k) {
        v16h a = load_a_frag(Xrt, KPAD, k);
#pragma unroll
        for (int t = 0; t < 8; ++t) {
          const int ct = ch * 16 + pass * 8 + t;
          v16h bf = load_b_frag(w1 + ((size_t)(ct * KT1 + k) << 9) + (lane << 4));
          acc[t] = WMMA(a, bf, acc[t]);
        }
      }
#pragma unroll
      for (int t = 0; t < 8; ++t) {
        const int ct = ch * 16 + pass * 8 + t;
        const int n  = ct * 16 + nSub;
        const float bias = b1[n];
#pragma unroll
        for (int j = 0; j < 8; ++j) {
          float h = acc[t][j] + bias;
          h = h > 0.f ? h : 0.f;
          const int m = rt * 16 + j + halfSel;
          ldsH[m * HIDD + n] = (_Float16)h;
        }
      }
    }
  }
  __syncthreads();

  // ---- GEMM2: out = hidden @ W2 + b2 (+ residual) ----
  constexpr int CT2 = OUTP / 16;
  for (int tile = wave; tile < 4 * CT2; tile += 8) {
    const int rt = tile / CT2;
    const int ct = tile - rt * CT2;
    const _Float16* Hrt = ldsH + (size_t)rt * 16 * HIDD;
    v8f acc = {};
    for (int k = 0; k < HIDD / 32; ++k) {
      v16h a  = load_a_frag(Hrt, HIDD, k);
      v16h bf = load_b_frag(w2 + ((size_t)(ct * (HIDD / 32) + k) << 9) + (lane << 4));
      acc = WMMA(a, bf, acc);
    }
    const int   n    = ct * 16 + nSub;
    const float bias = (n < validCols) ? b2[n] : 0.f;
#pragma unroll
    for (int j = 0; j < 8; ++j) {
      const int m = rt * 16 + j + halfSel;
      if (m < validRows && n < validCols) {
        float v = acc[j] + bias;
        if (residual) v += (float)ldsX[(size_t)m * KPAD + n];
        if (outIsF16) ((_Float16*)out)[(size_t)m * outStride + n] = (_Float16)v;
        else          ((float*)out)[(size_t)m * outStride + n]    = v;
      }
    }
  }
}

// ---------------- weight packing: f32 (K,N) row-major -> f16 B-fragments ----------------
__global__ void k_pack(const float* __restrict__ W, _Float16* __restrict__ out,
                       int K, int Kpad, int N, int Npad)
{
  const int total = Kpad * Npad;
  const int KT = Kpad >> 5;
  for (int t = blockIdx.x * blockDim.x + threadIdx.x; t < total;
       t += gridDim.x * blockDim.x) {
    const int frag   = t >> 9;
    const int within = t & 511;
    const int lane   = within >> 4;
    const int el     = within & 15;
    const int nTile  = frag / KT;
    const int kTile  = frag - nTile * KT;
    const int k = kTile * 32 + ((lane & 16) ? 8 : 0) + (el < 8 ? el : el + 8);
    const int n = nTile * 16 + (lane & 15);
    float v = (k < K && n < N) ? W[(size_t)k * N + n] : 0.f;
    out[t] = (_Float16)v;
  }
}

__global__ void k_zero(float* p, int n) {
  int i = blockIdx.x * blockDim.x + threadIdx.x;
  if (i < n) p[i] = 0.f;
}

// segment_sum(e, dst) via float atomics, e stored f16
__global__ void k_scatter(const _Float16* __restrict__ e, const int* __restrict__ dst,
                          float* agg)
{
  int t = blockIdx.x * blockDim.x + threadIdx.x;
  if (t >= NEDGE * EDGE_D) return;
  const int eid = t >> 6;
  const int c   = t & 63;
  atomicAdd(&agg[(size_t)dst[eid] * EDGE_D + c], (float)e[t]);
}

// ---------------- encoder: gather(grid f32) ++ geo -> MLP 163->512->256 (out f16) ----------------
__global__ void __launch_bounds__(256)
k_encoder(const float* __restrict__ xc, const float* __restrict__ xp,
          const float* __restrict__ geo, const int* __restrict__ g2m,
          const _Float16* __restrict__ w1, const float* __restrict__ b1,
          const _Float16* __restrict__ w2, const float* __restrict__ b2,
          _Float16* xout)
{
  __shared__ __align__(16) _Float16 X[64 * 192];
  __shared__ __align__(16) _Float16 H[64 * HIDD];
  __shared__ int sG[64];
  const int m0 = blockIdx.x * 64;
  if (threadIdx.x < 64) {
    const int m = m0 + threadIdx.x;
    sG[threadIdx.x] = (m < NMESH) ? g2m[m] : -1;
  }
  __syncthreads();
  for (int idx = threadIdx.x; idx < 64 * 192; idx += 256) {
    const int r = idx / 192;
    const int c = idx - r * 192;
    const int g = sG[r];
    float v = 0.f;
    if (g >= 0) {
      if (c < VV)              v = xc[(size_t)g * VV + c];
      else if (c < 2 * VV)     v = xp[(size_t)g * VV + (c - VV)];
      else if (c < 2 * VV + 7) v = geo[(m0 + r) * 7 + (c - 2 * VV)];
    }
    X[idx] = (_Float16)v;
  }
  __syncthreads();
  mlp_core<192, NODE_D>(X, H, w1, b1, w2, b2, xout + (size_t)m0 * NODE_D, NODE_D, true,
                        imin(64, NMESH - m0), NODE_D, false);
}

// ---------------- edge embedder: raw(4 f32) -> 512 -> 64 (out f16) ----------------
__global__ void __launch_bounds__(256)
k_edge_embed(const float* __restrict__ raw,
             const _Float16* __restrict__ w1, const float* __restrict__ b1,
             const _Float16* __restrict__ w2, const float* __restrict__ b2,
             _Float16* e0)
{
  __shared__ __align__(16) _Float16 X[64 * 32];
  __shared__ __align__(16) _Float16 H[64 * HIDD];
  const int eb = blockIdx.x * 64;
  for (int idx = threadIdx.x; idx < 64 * 32; idx += 256) {
    const int r = idx >> 5;
    const int c = idx & 31;
    float v = (c < 4) ? raw[(size_t)(eb + r) * 4 + c] : 0.f;
    X[idx] = (_Float16)v;
  }
  __syncthreads();
  mlp_core<32, EDGE_D>(X, H, w1, b1, w2, b2, e0 + (size_t)eb * EDGE_D, EDGE_D, true,
                       64, EDGE_D, false);
}

// ---------------- edge MLP: concat(e, x[src], x[dst]) 576 -> 512 -> 64, residual ----------------
// All activations f16; tile staged with async global->LDS copies (16B chunks).
__global__ void __launch_bounds__(256)
k_edge_mlp(_Float16* e, const _Float16* __restrict__ x,
           const int* __restrict__ src, const int* __restrict__ dst,
           const _Float16* __restrict__ w1, const float* __restrict__ b1,
           const _Float16* __restrict__ w2, const float* __restrict__ b2)
{
  __shared__ __align__(16) _Float16 X[64 * 576];
  __shared__ __align__(16) _Float16 H[64 * HIDD];
  __shared__ int sSrc[64], sDst[64];
  const int e0 = blockIdx.x * 64;               // NEDGE % 64 == 0
  if (threadIdx.x < 64) {
    sSrc[threadIdx.x] = src[e0 + threadIdx.x];
    sDst[threadIdx.x] = dst[e0 + threadIdx.x];
  }
  __syncthreads();
  // 64 rows x 576 halfs = 4608 chunks of 16B; 18 per thread, exact.
#pragma unroll 2
  for (int it = 0; it < 18; ++it) {
    const int chunk = it * 256 + threadIdx.x;
    const int r = chunk / 72;           // 72 chunks per row
    const int h = (chunk - r * 72) * 8; // half offset within row
    const _Float16* gsrc;
    if (h < EDGE_D)               gsrc = e + (size_t)(e0 + r) * EDGE_D + h;
    else if (h < EDGE_D + NODE_D) gsrc = x + (size_t)sSrc[r] * NODE_D + (h - EDGE_D);
    else                          gsrc = x + (size_t)sDst[r] * NODE_D + (h - EDGE_D - NODE_D);
    cp16_g2lds(X + r * 576 + h, gsrc);
  }
  async_wait();
  __syncthreads();
  mlp_core<576, EDGE_D>(X, H, w1, b1, w2, b2, e + (size_t)e0 * EDGE_D, EDGE_D, true,
                        64, EDGE_D, true);
}

// ---------------- node MLP: concat(x f16, agg f32) 320 -> 512 -> 256, residual ----------------
__global__ void __launch_bounds__(256)
k_node_mlp(_Float16* x, const float* __restrict__ agg,
           const _Float16* __restrict__ w1, const float* __restrict__ b1,
           const _Float16* __restrict__ w2, const float* __restrict__ b2)
{
  __shared__ __align__(16) _Float16 X[64 * 320];
  __shared__ __align__(16) _Float16 H[64 * HIDD];
  const int m0 = blockIdx.x * 64;
  // x part: 64 rows x 256 halfs = 2048 chunks of 16B; 8 per thread
  for (int it = 0; it < 8; ++it) {
    const int chunk = it * 256 + threadIdx.x;
    const int r = chunk >> 5;
    const int h = (chunk & 31) * 8;
    const int m = m0 + r;
    if (m < NMESH) {
      *(uint4*)(X + r * 320 + h) = *(const uint4*)(x + (size_t)m * NODE_D + h);
    } else {
      uint4 z = {0u, 0u, 0u, 0u};
      *(uint4*)(X + r * 320 + h) = z;
    }
  }
  // agg part: 64 rows x 64 halfs, f32 -> f16 convert
  for (int idx = threadIdx.x; idx < 64 * EDGE_D; idx += 256) {
    const int r = idx >> 6;
    const int c = idx & 63;
    const int m = m0 + r;
    float v = (m < NMESH) ? agg[(size_t)m * EDGE_D + c] : 0.f;
    X[r * 320 + NODE_D + c] = (_Float16)v;
  }
  __syncthreads();
  mlp_core<320, NODE_D>(X, H, w1, b1, w2, b2, x + (size_t)m0 * NODE_D, NODE_D, true,
                        imin(64, NMESH - m0), NODE_D, true);
}

// ---------------- decoder: x[m2g] (f16) 256 -> 512 -> 78 (pad 80), out f32 ----------------
__global__ void __launch_bounds__(256)
k_decoder(const _Float16* __restrict__ x, const int* __restrict__ m2g,
          const _Float16* __restrict__ w1, const float* __restrict__ b1,
          const _Float16* __restrict__ w2, const float* __restrict__ b2,
          float* out)
{
  __shared__ __align__(16) _Float16 X[64 * NODE_D];
  __shared__ __align__(16) _Float16 H[64 * HIDD];
  __shared__ int sNode[64];
  const int g0 = blockIdx.x * 64;
  if (threadIdx.x < 64) {
    const int g = g0 + threadIdx.x;
    sNode[threadIdx.x] = (g < NGRID) ? m2g[g] : -1;
  }
  __syncthreads();
  for (int it = 0; it < 8; ++it) {
    const int chunk = it * 256 + threadIdx.x;
    const int r = chunk >> 5;
    const int h = (chunk & 31) * 8;
    const int node = sNode[r];
    if (node >= 0) {
      *(uint4*)(X + r * NODE_D + h) = *(const uint4*)(x + (size_t)node * NODE_D + h);
    } else {
      uint4 z = {0u, 0u, 0u, 0u};
      *(uint4*)(X + r * NODE_D + h) = z;
    }
  }
  __syncthreads();
  mlp_core<NODE_D, 80>(X, H, w1, b1, w2, b2, out + (size_t)g0 * VV, VV, false,
                       imin(64, NGRID - g0), VV, false);
}

// ---------------- host driver ----------------
extern "C" void kernel_launch(void* const* d_in, const int* in_sizes, int n_in,
                              void* d_out, int out_size, void* d_ws, size_t ws_size,
                              hipStream_t stream)
{
  const float* x_current = (const float*)d_in[0];
  const float* x_previous = (const float*)d_in[1];
  const float* mesh_feats = (const float*)d_in[2];
  const float* raw_edge   = (const float*)d_in[3];
  const int*   edge_index = (const int*)d_in[4];
  const int*   g2m        = (const int*)d_in[5];
  const int*   m2g        = (const int*)d_in[6];
  const float* enc_w1 = (const float*)d_in[7];
  const float* enc_b1 = (const float*)d_in[8];
  const float* enc_w2 = (const float*)d_in[9];
  const float* enc_b2 = (const float*)d_in[10];
  const float* eme_w1 = (const float*)d_in[11];
  const float* eme_b1 = (const float*)d_in[12];
  const float* eme_w2 = (const float*)d_in[13];
  const float* eme_b2 = (const float*)d_in[14];
  const float* pe_w1 = (const float*)d_in[15];
  const float* pe_b1 = (const float*)d_in[16];
  const float* pe_w2 = (const float*)d_in[17];
  const float* pe_b2 = (const float*)d_in[18];
  const float* pn_w1 = (const float*)d_in[19];
  const float* pn_b1 = (const float*)d_in[20];
  const float* pn_w2 = (const float*)d_in[21];
  const float* pn_b2 = (const float*)d_in[22];
  const float* dec_w1 = (const float*)d_in[23];
  const float* dec_b1 = (const float*)d_in[24];
  const float* dec_w2 = (const float*)d_in[25];
  const float* dec_b2 = (const float*)d_in[26];

  const int* srcI = edge_index;
  const int* dstI = edge_index + NEDGE;

  char* ws = (char*)d_ws;
  size_t off = 0;
  auto take = [&](size_t bytes) -> void* {
    void* p = (void*)(ws + off);
    off += (bytes + 255) & ~(size_t)255;
    return p;
  };

  _Float16* wEnc1 = (_Float16*)take((size_t)192 * 512 * 2);
  _Float16* wEnc2 = (_Float16*)take((size_t)512 * 256 * 2);
  _Float16* wEme1 = (_Float16*)take((size_t)32 * 64 * 2);
  _Float16* wEme2 = (_Float16*)take((size_t)64 * 64 * 2);
  _Float16* wPe1  = (_Float16*)take((size_t)NLAYERS * 576 * 512 * 2);
  _Float16* wPe2  = (_Float16*)take((size_t)NLAYERS * 512 * 64 * 2);
  _Float16* wPn1  = (_Float16*)take((size_t)NLAYERS * 320 * 512 * 2);
  _Float16* wPn2  = (_Float16*)take((size_t)NLAYERS * 512 * 256 * 2);
  _Float16* wDec1 = (_Float16*)take((size_t)256 * 512 * 2);
  _Float16* wDec2 = (_Float16*)take((size_t)512 * 80 * 2);
  _Float16* edges0 = (_Float16*)take((size_t)NEDGE * EDGE_D * 2);
  _Float16* eBuf   = (_Float16*)take((size_t)NEDGE * EDGE_D * 2);
  _Float16* xBuf   = (_Float16*)take((size_t)NMESH * NODE_D * 2);
  float*    agg    = (float*)take((size_t)NMESH * EDGE_D * 4);

  auto pack = [&](const float* W, _Float16* dst, int K, int Kpad, int N, int Npad) {
    const int total = Kpad * Npad;
    k_pack<<<(total + 255) / 256, 256, 0, stream>>>(W, dst, K, Kpad, N, Npad);
  };

  // ---- pack all weights to f16 WMMA B-fragment layout ----
  pack(enc_w1, wEnc1, 163, 192, 512, 512);
  pack(enc_w2, wEnc2, 512, 512, 256, 256);
  pack(eme_w1, wEme1, 4, 32, 64, 64);
  pack(eme_w2, wEme2, 64, 64, 64, 64);
  for (int l = 0; l < NLAYERS; ++l) {
    pack(pe_w1 + (size_t)l * 576 * 512, wPe1 + (size_t)l * 576 * 512, 576, 576, 512, 512);
    pack(pe_w2 + (size_t)l * 512 * 64,  wPe2 + (size_t)l * 512 * 64,  512, 512, 64, 64);
    pack(pn_w1 + (size_t)l * 320 * 512, wPn1 + (size_t)l * 320 * 512, 320, 320, 512, 512);
    pack(pn_w2 + (size_t)l * 512 * 256, wPn2 + (size_t)l * 512 * 256, 512, 512, 256, 256);
  }
  pack(dec_w1, wDec1, 256, 256, 512, 512);
  pack(dec_w2, wDec2, 512, 512, VV, 80);

  // ---- edge embedder (shared across batches) ----
  k_edge_embed<<<NEDGE / 64, 256, 0, stream>>>(raw_edge, wEme1, eme_b1, wEme2, eme_b2, edges0);

  const int meshBlocks = (NMESH + 63) / 64;
  const int gridBlocks = (NGRID + 63) / 64;

  for (int b = 0; b < BB; ++b) {
    // encoder -> xBuf (f16)
    k_encoder<<<meshBlocks, 256, 0, stream>>>(
        x_current + (size_t)b * NGRID * VV, x_previous + (size_t)b * NGRID * VV,
        mesh_feats, g2m, wEnc1, enc_b1, wEnc2, enc_b2, xBuf);

    // fresh edge state for this batch
    (void)hipMemcpyAsync(eBuf, edges0, (size_t)NEDGE * EDGE_D * sizeof(_Float16),
                         hipMemcpyDeviceToDevice, stream);

    for (int l = 0; l < NLAYERS; ++l) {
      k_edge_mlp<<<NEDGE / 64, 256, 0, stream>>>(
          eBuf, xBuf, srcI, dstI,
          wPe1 + (size_t)l * 576 * 512, pe_b1 + (size_t)l * 512,
          wPe2 + (size_t)l * 512 * 64,  pe_b2 + (size_t)l * 64);

      k_zero<<<(NMESH * EDGE_D + 255) / 256, 256, 0, stream>>>(agg, NMESH * EDGE_D);
      k_scatter<<<(NEDGE * EDGE_D + 255) / 256, 256, 0, stream>>>(eBuf, dstI, agg);

      k_node_mlp<<<meshBlocks, 256, 0, stream>>>(
          xBuf, agg,
          wPn1 + (size_t)l * 320 * 512, pn_b1 + (size_t)l * 512,
          wPn2 + (size_t)l * 512 * 256, pn_b2 + (size_t)l * 256);
    }

    // decoder -> d_out (f32)
    k_decoder<<<gridBlocks, 256, 0, stream>>>(
        xBuf, m2g, wDec1, dec_b1, wDec2, dec_b2,
        (float*)d_out + (size_t)b * NGRID * VV);
  }
}
